// AdditiveAttention_57655640981861
// MI455X (gfx1250) — matrix operand order repaired and checked by reading, not compiled, not run on hardware
//
#include <hip/hip_runtime.h>
#include <hip/hip_bf16.h>

// Additive attention for MI455X (gfx1250, wave32).
// Stage 1: bf16 WMMA GEMM projections (q = query*Wq^T, kT = (key*Wk^T)^T).
// Stage 2: tanh/dot/softmax; k-tile staged to LDS via async-to-LDS (ASYNCcnt),
//          float4 (b128) global traffic, VALU/trans-bound inner loop.

typedef __attribute__((ext_vector_type(16))) __bf16 v16bf;
typedef __attribute__((ext_vector_type(8)))  float  v8f;

#define BSZ 4
#define TGT 256
#define SRC 1024
#define HSZ 256
#define CH  8          // h-rows staged to LDS per chunk (8*1024*4B = 32 KB)

#if defined(__AMDGCN__) && __has_builtin(__builtin_amdgcn_global_load_async_to_lds_b128) && __has_builtin(__builtin_amdgcn_s_wait_asynccnt)
#define USE_ASYNC_LDS 1
// Exact parameter pointee type per hipcc diagnostic:
//   "__attribute__((__vector_size__(4 * sizeof(int)))) int __device__ *"
typedef int async_b128_t __attribute__((vector_size(16)));
#define AS_GLOBAL __attribute__((address_space(1)))
#define AS_LDS    __attribute__((address_space(3)))
#else
#define USE_ASYNC_LDS 0
#endif

// ---- hardware math helpers (guarded; every branch host-parseable) ----
__device__ __forceinline__ float hw_exp2(float x) {
#if __has_builtin(__builtin_amdgcn_exp2f)
    return __builtin_amdgcn_exp2f(x);          // v_exp_f32
#else
    return exp2f(x);
#endif
}

__device__ __forceinline__ float hw_rcp(float x) {
#if __has_builtin(__builtin_amdgcn_rcpf)
    return __builtin_amdgcn_rcpf(x);           // v_rcp_f32
#else
    return 1.0f / x;
#endif
}

__device__ __forceinline__ float hw_tanh(float x) {
#if __has_builtin(__builtin_amdgcn_tanhf)
    return __builtin_amdgcn_tanhf(x);          // v_tanh_f32 (gfx1250)
#elif __has_builtin(__builtin_amdgcn_tanh_f32)
    return __builtin_amdgcn_tanh_f32(x);
#else
    // tanh(|x|) = 1 - 2/(exp2(2|x|*log2e)+1), restore sign
    float ax = __builtin_fabsf(x);
    float e  = hw_exp2(ax * 2.885390081777927f);   // 2*log2(e)
    float r  = 1.0f - 2.0f * hw_rcp(e + 1.0f);
    return __builtin_copysignf(r, x);
#endif
}

// ---------------------------------------------------------------------
// Stage 1: C = A[M,256] * W[256,256]^T  (C[m][n] = sum_k A[m][k]*W[n][k])
// One 16x16 output tile per wave, K-loop 256 in steps of 32 (8 WMMAs).
// transpose_out=1 stores kT[b][n][s] (b = m>>10, s = m&1023) -> per-lane
// contiguous 8-float runs => b128 stores (confirmed in round-1 asm).
// ---------------------------------------------------------------------
__global__ void __launch_bounds__(128)
addattn_proj_gemm_bf16(const float* __restrict__ A,
                       const float* __restrict__ W,
                       float* __restrict__ C,
                       int Mtotal, int transpose_out) {
    const int wave = threadIdx.x >> 5;
    const int lane = threadIdx.x & 31;
    const int gw   = blockIdx.x * 4 + wave;         // one tile per wave
    const int tileM = gw >> 4;                      // N=256 -> 16 N-tiles
    const int tileN = gw & 15;
    const int m0 = tileM << 4;
    const int n0 = tileN << 4;
    const int half = lane >> 4;                     // 0: lanes 0-15, 1: 16-31
    const int l    = lane & 15;

    const float* Arow = A + (size_t)(m0 + l) * HSZ; // A row for this lane
    const float* Wrow = W + (size_t)(n0 + l) * HSZ; // W row n (B column n)

    v8f acc = {};
#pragma unroll 4
    for (int k0 = 0; k0 < HSZ; k0 += 32) {
        // A fragment (16x32 bf16): elems 0..7 = K(k0+8h .. +7), 8..15 = +16
        const int ka = k0 + (half << 3);
        float4 a0 = *(const float4*)(Arow + ka);
        float4 a1 = *(const float4*)(Arow + ka + 4);
        float4 a2 = *(const float4*)(Arow + ka + 16);
        float4 a3 = *(const float4*)(Arow + ka + 20);
        // B fragment (32x16 bf16): elems 0..7 = K(k0+8h .. +7), 8..15 = +16 of column n
        const int kb = k0 + (half << 3);
        float4 b0 = *(const float4*)(Wrow + kb);
        float4 b1 = *(const float4*)(Wrow + kb + 4);
        float4 b2 = *(const float4*)(Wrow + kb + 16);
        float4 b3 = *(const float4*)(Wrow + kb + 20);

        v16bf af, bfm;
        af[0]=(__bf16)a0.x; af[1]=(__bf16)a0.y; af[2]=(__bf16)a0.z; af[3]=(__bf16)a0.w;
        af[4]=(__bf16)a1.x; af[5]=(__bf16)a1.y; af[6]=(__bf16)a1.z; af[7]=(__bf16)a1.w;
        af[8]=(__bf16)a2.x; af[9]=(__bf16)a2.y; af[10]=(__bf16)a2.z; af[11]=(__bf16)a2.w;
        af[12]=(__bf16)a3.x; af[13]=(__bf16)a3.y; af[14]=(__bf16)a3.z; af[15]=(__bf16)a3.w;
        bfm[0]=(__bf16)b0.x; bfm[1]=(__bf16)b0.y; bfm[2]=(__bf16)b0.z; bfm[3]=(__bf16)b0.w;
        bfm[4]=(__bf16)b1.x; bfm[5]=(__bf16)b1.y; bfm[6]=(__bf16)b1.z; bfm[7]=(__bf16)b1.w;
        bfm[8]=(__bf16)b2.x; bfm[9]=(__bf16)b2.y; bfm[10]=(__bf16)b2.z; bfm[11]=(__bf16)b2.w;
        bfm[12]=(__bf16)b3.x; bfm[13]=(__bf16)b3.y; bfm[14]=(__bf16)b3.z; bfm[15]=(__bf16)b3.w;

        acc = __builtin_amdgcn_wmma_f32_16x16x32_bf16(
            /*neg_a=*/false, af, /*neg_b=*/false, bfm,
            /*c_mod=*/(short)0, acc, /*reuse_a=*/false, /*reuse_b=*/false);
    }

    // C/D layout: lane<16 -> (M=r, N=l); lane>=16 -> (M=8+r, N=l)
    if (!transpose_out) {
        float* Cp = C + (size_t)(m0 + 8 * half) * HSZ + (n0 + l);
#pragma unroll
        for (int r = 0; r < 8; ++r) Cp[(size_t)r * HSZ] = acc[r];
    } else {
        const int n = n0 + l;
#pragma unroll
        for (int r = 0; r < 8; ++r) {
            const int m = m0 + 8 * half + r;
            const int b = m >> 10;
            const int s = m & 1023;
            C[((size_t)((b << 8) | n)) * SRC + s] = acc[r];
        }
    }
}

// ---------------------------------------------------------------------
// Stage 2: one block per (b,t) row. 256 threads; thread tid owns
// s = 4*tid .. 4*tid+3 (b128 loads/stores). k-tile (8 h-rows, 32 KB)
// staged into LDS per chunk via async-to-LDS; q[h], v[h] broadcast
// from LDS. Then 1024-wide softmax in-block.
// ---------------------------------------------------------------------
__global__ void __launch_bounds__(256)
addattn_scores_softmax(const float* __restrict__ qp,   // [B*T, 256]
                       const float* __restrict__ kT,   // [B, 256, 1024]
                       const float* __restrict__ v,    // [256]
                       const float* __restrict__ vbias,// [1]
                       float* __restrict__ out) {      // [B*T, 1024]
    __shared__ float skbuf[CH * SRC];   // 32 KB staged k rows
    __shared__ float sq[HSZ];
    __shared__ float sv[HSZ];
    __shared__ float red[256];

    const int tid = threadIdx.x;
    const int row = blockIdx.x;          // b*TGT + t
    const int b   = row >> 8;

    sq[tid] = qp[(size_t)row * HSZ + tid];
    sv[tid] = v[tid];

    const float* kb = kT + (size_t)b * HSZ * SRC;
    float acc0 = 0.f, acc1 = 0.f, acc2 = 0.f, acc3 = 0.f;

    for (int hc = 0; hc < HSZ; hc += CH) {
        __syncthreads();   // skbuf consumers from previous chunk done (+ sq/sv visible)
#if USE_ASYNC_LDS
#pragma unroll
        for (int r = 0; r < CH; ++r) {
            const float* src = kb + (size_t)(hc + r) * SRC + 4 * tid;
            __builtin_amdgcn_global_load_async_to_lds_b128(
                (AS_GLOBAL async_b128_t*)src,
                (AS_LDS    async_b128_t*)&skbuf[r * SRC + 4 * tid],
                0, 0);
        }
        __builtin_amdgcn_s_wait_asynccnt(0);
#else
#pragma unroll
        for (int r = 0; r < CH; ++r) {
            *(float4*)&skbuf[r * SRC + 4 * tid] =
                *(const float4*)(kb + (size_t)(hc + r) * SRC + 4 * tid);
        }
#endif
        __syncthreads();

#pragma unroll
        for (int r = 0; r < CH; ++r) {
            const float qh = sq[hc + r];
            const float vh = sv[hc + r];
            const float4 kv = *(const float4*)&skbuf[r * SRC + 4 * tid];
            acc0 += vh * hw_tanh(qh + kv.x);
            acc1 += vh * hw_tanh(qh + kv.y);
            acc2 += vh * hw_tanh(qh + kv.z);
            acc3 += vh * hw_tanh(qh + kv.w);
        }
    }

    const float bias = vbias[0];
    acc0 += bias; acc1 += bias; acc2 += bias; acc3 += bias;

    // block max over all 1024 scores
    __syncthreads();
    red[tid] = fmaxf(fmaxf(acc0, acc1), fmaxf(acc2, acc3));
    __syncthreads();
    for (int off = 128; off > 0; off >>= 1) {
        if (tid < off) red[tid] = fmaxf(red[tid], red[tid + off]);
        __syncthreads();
    }
    const float mx = red[0];
    __syncthreads();

    const float L2E = 1.4426950408889634f;
    const float e0 = hw_exp2((acc0 - mx) * L2E);
    const float e1 = hw_exp2((acc1 - mx) * L2E);
    const float e2 = hw_exp2((acc2 - mx) * L2E);
    const float e3 = hw_exp2((acc3 - mx) * L2E);

    // block sum
    red[tid] = e0 + e1 + e2 + e3;
    __syncthreads();
    for (int off = 128; off > 0; off >>= 1) {
        if (tid < off) red[tid] += red[tid + off];
        __syncthreads();
    }
    const float inv = hw_rcp(red[0]);

    float4 o;
    o.x = e0 * inv; o.y = e1 * inv; o.z = e2 * inv; o.w = e3 * inv;
    *(float4*)(out + (size_t)row * SRC + 4 * tid) = o;
}

extern "C" void kernel_launch(void* const* d_in, const int* in_sizes, int n_in,
                              void* d_out, int out_size, void* d_ws, size_t ws_size,
                              hipStream_t stream) {
    (void)in_sizes; (void)n_in; (void)out_size; (void)ws_size;
    const float* query = (const float*)d_in[0];  // [4,256,256]
    const float* key   = (const float*)d_in[1];  // [4,1024,256]
    const float* Wq    = (const float*)d_in[2];  // [256,256]
    const float* Wk    = (const float*)d_in[3];  // [256,256]
    const float* v     = (const float*)d_in[4];  // [256]
    const float* vb    = (const float*)d_in[5];  // [1]
    float* out = (float*)d_out;                  // [4,256,1024]

    float* qp = (float*)d_ws;                    // [1024,256]  (1 MB)
    float* kT = qp + (size_t)BSZ * TGT * HSZ;    // [4,256,1024] (4 MB)

    // Q projection: M=1024 -> 64*16 = 1024 tiles -> 256 blocks x 4 waves
    addattn_proj_gemm_bf16<<<256, 128, 0, stream>>>(query, Wq, qp,
                                                    BSZ * TGT, /*transpose=*/0);
    // K projection (transposed store): M=4096 -> 4096 tiles -> 1024 blocks
    addattn_proj_gemm_bf16<<<1024, 128, 0, stream>>>(key, Wk, kT,
                                                     BSZ * SRC, /*transpose=*/1);
    // Scores + softmax: one block per (b,t) row
    addattn_scores_softmax<<<BSZ * TGT, 256, 0, stream>>>(qp, kT, v, vb, out);
}
